// SelfAttention_2362232012993
// MI455X (gfx1250) — matrix operand order repaired and checked by reading, not compiled, not run on hardware
//
#include <hip/hip_runtime.h>
#include <hip/hip_bf16.h>

// ---------------------------------------------------------------------------
// Multi-head self-attention for MI455X (gfx1250, wave32, WMMA).
// Fixed problem size: N=2, L=2048, E=1024, H=16, D=64.
// All GEMMs via v_wmma_f32_16x16x32_f16 (f16 in, f32 accumulate).
// K/V tiles staged into LDS with GLOBAL_LOAD_ASYNC_TO_LDS_B128 (ASYNCcnt).
// Flash-style online softmax kept in registers (half-wave shfl reductions).
// ---------------------------------------------------------------------------

typedef __attribute__((ext_vector_type(16))) _Float16 v16h;
typedef __attribute__((ext_vector_type(8)))  float    v8f;
typedef _Float16 h16;

#define NB     2
#define SL     2048
#define EMB    1024
#define HEADS  16
#define HD     64
#define SCALE  0.03125f          // 1/sqrt(1024)  (reference scales by sqrt(E))
#define NEGBIG (-3.125e18f)      // (-1e20) * SCALE

static __device__ __forceinline__ v8f wmma_f16(v16h a, v16h b, v8f c) {
    return __builtin_amdgcn_wmma_f32_16x16x32_f16(
        /*neg_a=*/false, a, /*neg_b=*/false, b,
        /*c_mod=*/(short)0, c, /*reuse_a=*/false, /*reuse_b=*/false);
}

static __device__ __forceinline__ void sched_fence() {
#if __has_builtin(__builtin_amdgcn_sched_barrier)
    __builtin_amdgcn_sched_barrier(0);
#endif
}

// Async DMA: 16 bytes global -> LDS per lane, tracked by ASYNCcnt.
// VDST holds the LDS byte address (low 32 bits of a generic LDS pointer),
// VADDR holds the 64-bit global address.
static __device__ __forceinline__ void async_load_b128(void* lds_ptr, const void* gptr) {
    unsigned lo = (unsigned)(size_t)lds_ptr;               // LDS byte offset
    unsigned long long ga = (unsigned long long)(size_t)gptr;
    asm volatile("global_load_async_to_lds_b128 %0, %1, off"
                 :: "v"(lo), "v"(ga) : "memory");
}
static __device__ __forceinline__ void wait_async0() {
    asm volatile("s_wait_asynccnt 0" ::: "memory");
}

// A-matrix 16x32 f16 fragment (ISA 7.12.2 16-bit A layout):
//   lane<16 : row = lane,    halves 0..7 -> k0+0..7,   halves 8..15 -> k0+16..23
//   lane>=16: row = lane-16, halves 0..7 -> k0+8..15,  halves 8..15 -> k0+24..31
static __device__ __forceinline__ v16h load_a_f16(const h16* base, int ldm,
                                                  int row0, int k0, int lane) {
    int r  = row0 + (lane & 15);
    int kb = k0 + ((lane & 16) ? 8 : 0);
    const h16* p = base + (size_t)r * ldm + kb;
    union { v16h v; uint4 q[2]; } u;
    u.q[0] = *(const uint4*)(p);        // halves 0..7  (16B)
    u.q[1] = *(const uint4*)(p + 16);   // halves 8..15 (16B)
    return u.v;
}

// Same A fragment but sourced from fp32 memory (convert on load).
static __device__ __forceinline__ v16h load_a_f32(const float* base, int ldm,
                                                  int row0, int k0, int lane) {
    int r  = row0 + (lane & 15);
    int kb = k0 + ((lane & 16) ? 8 : 0);
    const float* p = base + (size_t)r * ldm + kb;
    union { v16h v; h16 h[16]; } u;
#pragma unroll
    for (int i = 0; i < 8; ++i) {
        u.h[i]     = (h16)p[i];
        u.h[8 + i] = (h16)p[16 + i];
    }
    return u.v;
}

// B-matrix 32x16 f16 fragment where B = S^T (B column n is row n of the
// row-major matrix S).  B layout: lane<16 holds K=0..7,16..23, lane>=16 K=8..15,24..31,
// lane%16 = column. -> two 16B reads of row n of S.
static __device__ __forceinline__ v16h load_bT_f16(const h16* base, int ldm,
                                                   int n0, int k0, int lane) {
    int n  = n0 + (lane & 15);
    int kb = k0 + ((lane & 16) ? 8 : 0);
    const h16* p = base + (size_t)n * ldm + kb;
    union { v16h v; uint4 q[2]; } u;
    u.q[0] = *(const uint4*)(p);
    u.q[1] = *(const uint4*)(p + 16);
    return u.v;
}

// B-matrix 32x16 fragment from row-major K-major memory (B[k][n] = base[k][n]):
// strided gather (used for V tiles staged in LDS).
static __device__ __forceinline__ v16h load_b_strided_f16(const h16* base, int ldm,
                                                          int k0, int n0, int lane) {
    int n  = n0 + (lane & 15);
    int kb = k0 + ((lane & 16) ? 8 : 0);
    union { v16h v; h16 h[16]; } u;
#pragma unroll
    for (int hh = 0; hh < 16; ++hh)
        u.h[hh] = base[(size_t)(kb + hh + (hh & 8)) * ldm + n];
    return u.v;
}

// ---------------------------------------------------------------------------
__global__ void cvt_f32_f16(const float* __restrict__ src, h16* __restrict__ dst, int n) {
    int i = blockIdx.x * blockDim.x + threadIdx.x;
    if (i < n) dst[i] = (h16)src[i];
}

// ---------------------------------------------------------------------------
// One tensor of the QKV projection: 16 rows x (64->64) for this wave.
static __device__ __forceinline__ void proj_one(const float* __restrict__ X,
                                                const h16* __restrict__ W,
                                                h16* __restrict__ O,
                                                int row0, int lane, int cr) {
    v16h a0 = load_a_f32(X, 64, row0, 0, lane);
    v16h a1 = load_a_f32(X, 64, row0, 32, lane);
#pragma unroll
    for (int nt = 0; nt < 4; ++nt) {
        v8f c = {};
        c = wmma_f16(a0, load_bT_f16(W, 64, nt * 16, 0, lane), c);
        c = wmma_f16(a1, load_bT_f16(W, 64, nt * 16, 32, lane), c);
        int col = nt * 16 + (lane & 15);
#pragma unroll
        for (int i = 0; i < 8; ++i)
            O[(size_t)(row0 + cr + i) * 64 + col] = (h16)c[i];
    }
}

// QKV projection: rows = N*L*H = 65536 of 64 contiguous fp32; out = x @ W^T.
// 128 threads = 4 waves; each wave does a 16-row strip, all 64 output cols.
// Tensors are processed strictly sequentially (sched fences) to avoid spills.
__global__ void qkv_proj(const float* __restrict__ q, const float* __restrict__ k,
                         const float* __restrict__ v,
                         const h16* __restrict__ WqH, const h16* __restrict__ WkH,
                         const h16* __restrict__ WvH,
                         h16* __restrict__ Qp, h16* __restrict__ Kp, h16* __restrict__ Vp) {
    int lane = threadIdx.x & 31;
    int wv   = threadIdx.x >> 5;
    int row0 = blockIdx.x * 64 + wv * 16;
    int cr   = (lane & 16) ? 8 : 0;

    proj_one(q, WqH, Qp, row0, lane, cr);
    sched_fence();
    proj_one(k, WkH, Kp, row0, lane, cr);
    sched_fence();
    proj_one(v, WvH, Vp, row0, lane, cr);
}

// ---------------------------------------------------------------------------
// Fused attention with online softmax.
// Grid: N*H*(L/64) = 1024 blocks, 128 threads (4 waves x 16 query rows).
// Each block owns one (n, head, 64-query-row) tile and streams keys in
// tiles of 32; K/V tiles go memory->LDS via async DMA (ASYNCcnt).
__global__ void attention(const h16* __restrict__ Qp, const h16* __restrict__ Kp,
                          const h16* __restrict__ Vp, const int* __restrict__ masks,
                          h16* __restrict__ Ctx) {
    __shared__ h16 Klds[32][72];           // 32 keys x 64 dims (+pad)
    __shared__ h16 Vlds[32][72];
    __shared__ h16 Plds[4][16][40];        // per-wave prob tile 16x32 (+pad)

    int lane = threadIdx.x & 31;
    int wv   = threadIdx.x >> 5;

    int b  = blockIdx.x;
    int qb = b & 31;                       // query block within sequence
    int hh = (b >> 5) & 15;                // head
    int n  = b >> 9;                       // batch

    int q0 = qb * 64 + wv * 16;            // this wave's first query row
    const h16* qbase = Qp + (size_t)n * SL * EMB + hh * HD;
    const h16* kbase = Kp + (size_t)n * SL * EMB + hh * HD;
    const h16* vbase = Vp + (size_t)n * SL * EMB + hh * HD;
    const int* mbase = masks + (size_t)n * SL;

    // Q fragments for both K-dim steps (D=64 -> 2 x 32)
    v16h qa0 = load_a_f16(qbase, EMB, q0, 0, lane);
    v16h qa1 = load_a_f16(qbase, EMB, q0, 32, lane);

    v8f o0 = {}, o1 = {}, o2 = {}, o3 = {};
    float M[8], Lsum[8];
#pragma unroll
    for (int i = 0; i < 8; ++i) { M[i] = -INFINITY; Lsum[i] = 0.0f; }

    int cr = (lane & 16) ? 8 : 0;          // C-layout row offset for this lane
    int lt = threadIdx.x;
    int lr = lt >> 2;                      // 0..31 : tile row loaded by thread
    int ls = (lt & 3) * 16;                // 0/16/32/48 : half offset

    for (int j0 = 0; j0 < SL; j0 += 32) {
        // ---- async DMA stage of K/V 32x64 tiles into LDS ----
        {
            const h16* ks = kbase + (size_t)(j0 + lr) * EMB + ls;
            const h16* vs = vbase + (size_t)(j0 + lr) * EMB + ls;
            async_load_b128(&Klds[lr][ls],     ks);
            async_load_b128(&Klds[lr][ls + 8], ks + 8);
            async_load_b128(&Vlds[lr][ls],     vs);
            async_load_b128(&Vlds[lr][ls + 8], vs + 8);
            if (j0 + 32 < SL) {            // prefetch next tile into L2
                __builtin_prefetch(ks + EMB * 32, 0, 1);
                __builtin_prefetch(vs + EMB * 32, 0, 1);
            }
        }
        wait_async0();
        __syncthreads();

        // ---- scores: S = Q * K^T for 32 keys (two 16-col C tiles) ----
        v8f c0 = {}, c1 = {};
        c0 = wmma_f16(qa0, load_bT_f16(&Klds[0][0], 72, 0, 0, lane), c0);
        c0 = wmma_f16(qa1, load_bT_f16(&Klds[0][0], 72, 0, 32, lane), c0);
        c1 = wmma_f16(qa0, load_bT_f16(&Klds[0][0], 72, 16, 0, lane), c1);
        c1 = wmma_f16(qa1, load_bT_f16(&Klds[0][0], 72, 16, 32, lane), c1);

        int m0 = mbase[j0 + (lane & 15)];
        int m1 = mbase[j0 + 16 + (lane & 15)];

        // ---- scale + mask + online softmax update ----
        float p0a[8], p1a[8], fa[8];
#pragma unroll
        for (int i = 0; i < 8; ++i) {
            float s0 = m0 ? c0[i] * SCALE : NEGBIG;
            float s1 = m1 ? c1[i] * SCALE : NEGBIG;
            float t = fmaxf(s0, s1);
            t = fmaxf(t, __shfl_xor(t, 1, 32));
            t = fmaxf(t, __shfl_xor(t, 2, 32));
            t = fmaxf(t, __shfl_xor(t, 4, 32));
            t = fmaxf(t, __shfl_xor(t, 8, 32));
            float newM = fmaxf(M[i], t);
            float corr = __expf(M[i] - newM);
            float p0 = __expf(s0 - newM);
            float p1 = __expf(s1 - newM);
            float ps = p0 + p1;
            ps += __shfl_xor(ps, 1, 32);
            ps += __shfl_xor(ps, 2, 32);
            ps += __shfl_xor(ps, 4, 32);
            ps += __shfl_xor(ps, 8, 32);
            Lsum[i] = Lsum[i] * corr + ps;
            M[i] = newM;
            fa[i] = corr; p0a[i] = p0; p1a[i] = p1;
        }
#pragma unroll
        for (int i = 0; i < 8; ++i) {
            o0[i] *= fa[i]; o1[i] *= fa[i]; o2[i] *= fa[i]; o3[i] *= fa[i];
        }

        // ---- reshape P (C layout -> A layout) via per-wave LDS tile ----
#pragma unroll
        for (int i = 0; i < 8; ++i) {
            Plds[wv][cr + i][lane & 15]        = (h16)p0a[i];
            Plds[wv][cr + i][16 + (lane & 15)] = (h16)p1a[i];
        }
        // LDS ops from the same wave execute in order; no barrier needed.
        v16h pf = load_a_f16(&Plds[wv][0][0], 40, 0, 0, lane);

        // ---- O += P * V (4 x 16-dim output tiles) ----
        o0 = wmma_f16(pf, load_b_strided_f16(&Vlds[0][0], 72, 0, 0,  lane), o0);
        o1 = wmma_f16(pf, load_b_strided_f16(&Vlds[0][0], 72, 0, 16, lane), o1);
        o2 = wmma_f16(pf, load_b_strided_f16(&Vlds[0][0], 72, 0, 32, lane), o2);
        o3 = wmma_f16(pf, load_b_strided_f16(&Vlds[0][0], 72, 0, 48, lane), o3);

        __syncthreads();   // before K/V LDS is overwritten
    }

    // ---- normalize and store context (f16, [N,L,H,D] layout) ----
    float inv[8];
#pragma unroll
    for (int i = 0; i < 8; ++i) inv[i] = 1.0f / Lsum[i];
    h16* cbase = Ctx + (size_t)n * SL * EMB + hh * HD;
    int colb = lane & 15;
#pragma unroll
    for (int i = 0; i < 8; ++i) {
        size_t r = (size_t)(q0 + cr + i) * EMB;
        cbase[r + 0  + colb] = (h16)(o0[i] * inv[i]);
        cbase[r + 16 + colb] = (h16)(o1[i] * inv[i]);
        cbase[r + 32 + colb] = (h16)(o2[i] * inv[i]);
        cbase[r + 48 + colb] = (h16)(o3[i] * inv[i]);
    }
}

// ---------------------------------------------------------------------------
// Output projection: out = ctx @ Wo^T + bo.  M=4096, K=N=1024.
// Grid 4096 blocks x 128 threads; each wave owns one 16x16 output tile.
__global__ void out_proj(const h16* __restrict__ Ctx, const h16* __restrict__ WoH,
                         const float* __restrict__ bo, float* __restrict__ out) {
    int lane = threadIdx.x & 31;
    int wv   = threadIdx.x >> 5;
    int b    = blockIdx.x;
    int rt   = b & 255;                    // 256 row tiles
    int cg   = b >> 8;                     // 16 col groups of 4 waves
    int row0 = rt * 16;
    int col0 = (cg * 4 + wv) * 16;

    v8f c = {};
#pragma unroll 4
    for (int ks = 0; ks < 32; ++ks) {
        v16h a  = load_a_f16(Ctx, EMB, row0, ks * 32, lane);
        v16h bf = load_bT_f16(WoH, EMB, col0, ks * 32, lane);
        c = wmma_f16(a, bf, c);
    }
    int col = col0 + (lane & 15);
    float bias = bo[col];
    int cr = (lane & 16) ? 8 : 0;
#pragma unroll
    for (int i = 0; i < 8; ++i)
        out[(size_t)(row0 + cr + i) * EMB + col] = c[i] + bias;
}

// ---------------------------------------------------------------------------
extern "C" void kernel_launch(void* const* d_in, const int* in_sizes, int n_in,
                              void* d_out, int out_size, void* d_ws, size_t ws_size,
                              hipStream_t stream) {
    (void)in_sizes; (void)n_in; (void)out_size;
    const float* query  = (const float*)d_in[0];
    const float* keys   = (const float*)d_in[1];
    const float* values = (const float*)d_in[2];
    const int*   masks  = (const int*)d_in[3];
    const float* Wq     = (const float*)d_in[4];
    const float* Wk     = (const float*)d_in[5];
    const float* Wv     = (const float*)d_in[6];
    const float* Wo     = (const float*)d_in[7];
    const float* bo     = (const float*)d_in[8];
    float* outp         = (float*)d_out;

    // Workspace layout (halves)
    h16* ws  = (h16*)d_ws;
    h16* WqH = ws;                         // 4096
    h16* WkH = WqH + 4096;                 // 4096
    h16* WvH = WkH + 4096;                 // 4096
    h16* WoH = WvH + 4096;                 // 1048576
    h16* Qp  = WoH + 1048576;              // N*L*E = 4194304
    h16* Kp  = Qp + (size_t)NB * SL * EMB;
    h16* Vp  = Kp + (size_t)NB * SL * EMB;
    h16* Ctx = Vp + (size_t)NB * SL * EMB;
    size_t need = ((size_t)(Ctx - ws) + (size_t)NB * SL * EMB) * sizeof(h16);
    if (ws_size < need) return;            // ~34 MB required

    cvt_f32_f16<<<(4096 + 255) / 256, 256, 0, stream>>>(Wq, WqH, 4096);
    cvt_f32_f16<<<(4096 + 255) / 256, 256, 0, stream>>>(Wk, WkH, 4096);
    cvt_f32_f16<<<(4096 + 255) / 256, 256, 0, stream>>>(Wv, WvH, 4096);
    cvt_f32_f16<<<(1048576 + 255) / 256, 256, 0, stream>>>(Wo, WoH, 1048576);

    qkv_proj<<<1024, 128, 0, stream>>>(query, keys, values, WqH, WkH, WvH, Qp, Kp, Vp);
    attention<<<1024, 128, 0, stream>>>(Qp, Kp, Vp, masks, Ctx);
    out_proj<<<4096, 128, 0, stream>>>(Ctx, WoH, bo, outp);
}
